// My_APPNP_62663572848797
// MI455X (gfx1250) — matrix operand order repaired and checked by reading, not compiled, not run on hardware
//
#include <hip/hip_runtime.h>
#include <hip/hip_bf16.h>

typedef float  v8f   __attribute__((ext_vector_type(8)));
typedef __bf16 v16bf __attribute__((ext_vector_type(16)));

#define N_FEAT 512
#define HID    256
#define COUT   64
#define WAVES  4
#define ROWS_PER_BLOCK (WAVES * 16)   // 64 rows per block
// LDS byte layout (64 KB total):
//   stage1:  [0, 4K)   A frags: 4 row-tiles x 1KB (current 32-K chunk)
//            [4K, 20K) B frags: 16 col-tiles x 1KB (current 32-K chunk)
//   stage2:  [0, 32K)  h1 A-frags: 4 waves x 8KB  (written after barrier)
//            [32K,64K) W2 B-frags: 8 K-chunks x 4 col-tiles x 1KB (written once)
#define LDS_B1  (4 * 1024)
#define LDS_W2  (32 * 1024)
#define LDS_SZ  (64 * 1024)

#define WMMA_BF16(A, B, C) \
  __builtin_amdgcn_wmma_f32_16x16x32_bf16(false, (A), false, (B), (short)0, (C), false, false)

__device__ __forceinline__ unsigned short f2bf(float f) {
  unsigned int u = __float_as_uint(f);
  u += 0x7FFFu + ((u >> 16) & 1u);     // round-to-nearest-even truncation
  return (unsigned short)(u >> 16);
}

// Byte offset of element (m, koff) inside a 1KB 16x32 bf16 A-fragment chunk.
// ISA 7.12.2: lanes 0-15 hold K in [0,8)u[16,24); lanes 16-31 hold [8,16)u[24,32).
__device__ __forceinline__ int afrag_off(int m, int koff) {
  int hi   = (koff >> 3) & 1;
  int i    = (koff & 7) + ((koff >> 4) & 1) * 8;
  int lane = (m & 15) + hi * 16;
  return lane * 32 + i * 2;
}

// Byte offset of element (koff, ncol) inside a 1KB 32x16 bf16 B-fragment chunk.
// lanes 0-15: K in [0,8)u[16,24), N=lane; lanes 16-31: K in [8,16)u[24,32), N=lane-16.
__device__ __forceinline__ int bfrag_off(int koff, int ncol) {
  int lane = (ncol & 15) + ((koff >> 3) & 1) * 16;
  int i    = (koff & 7) + ((koff >> 4) & 1) * 8;
  return lane * 32 + i * 2;
}

__global__ void __launch_bounds__(128)
encoder_kernel(const float* __restrict__ x,  const float* __restrict__ W1,
               const float* __restrict__ b1, const float* __restrict__ W2,
               const float* __restrict__ b2, float* __restrict__ h,
               float* __restrict__ z, int N) {
  extern __shared__ char smem[];
  const int tid  = threadIdx.x;
  const int lane = tid & 31;
  const int wave = tid >> 5;
  const int row0 = blockIdx.x * ROWS_PER_BLOCK;
  const int hi8  = (lane >> 4) * 8;
  const v8f vzero = {0.f, 0.f, 0.f, 0.f, 0.f, 0.f, 0.f, 0.f};

  // ---- One-time: stage W2 (256x64) as bf16 B-fragments at [32K,64K) ----
  for (int idx = tid; idx < HID * COUT; idx += 128) {
    int k = idx >> 6, c = idx & 63;
    *(unsigned short*)(smem + LDS_W2 + (k >> 5) * 4096 + (c >> 4) * 1024 +
                       bfrag_off(k & 31, c & 15)) = f2bf(W2[k * COUT + c]);
  }

  // ---- Stage 1: h1 = relu(x @ W1 + b1); each wave owns a 16x256 tile ----
  v8f acc[16];
#pragma unroll
  for (int n = 0; n < 16; ++n) acc[n] = vzero;

  for (int kk = 0; kk < N_FEAT / 32; ++kk) {
    __syncthreads();
    // stage A: 64 rows x 32 K (coalesced global reads, swizzled LDS writes)
    for (int idx = tid; idx < ROWS_PER_BLOCK * 32; idx += 128) {
      int m = idx >> 5, koff = idx & 31;
      int gr = row0 + m; if (gr >= N) gr = N - 1;          // clamp, keep EXEC full
      *(unsigned short*)(smem + (m >> 4) * 1024 + afrag_off(m & 15, koff)) =
          f2bf(x[(size_t)gr * N_FEAT + kk * 32 + koff]);
    }
    // stage B: 32 x 256 chunk of W1
    for (int idx = tid; idx < 32 * HID; idx += 128) {
      int koff = idx >> 8, c = idx & 255;
      *(unsigned short*)(smem + LDS_B1 + (c >> 4) * 1024 +
                         bfrag_off(koff, c & 15)) = f2bf(W1[(kk * 32 + koff) * HID + c]);
    }
    __syncthreads();

    // Prefetch next K-chunk of x and W1 into cache while this chunk computes.
    // Lowers to global_prefetch_b8: no VGPR return, no counter traffic.
    if (kk + 1 < N_FEAT / 32) {
      int m = tid >> 1;                                    // 2 threads per row, 64B each
      int gr = row0 + m; if (gr >= N) gr = N - 1;
      __builtin_prefetch(&x[(size_t)gr * N_FEAT + (kk + 1) * 32 + (tid & 1) * 16], 0, 3);
      __builtin_prefetch(&W1[((kk + 1) * 32 + (tid >> 2)) * HID + (tid & 3) * 64], 0, 3);
    }

    // Batched fragment consumption: 4 B-fragments live at once (4 distinct
    // VGPR octets), so 8 ds_load_b128 issue back-to-back and the dscnt waits
    // overlap the multicycle WMMAs instead of fencing each one.
    const v16bf a = *(const v16bf*)(smem + wave * 1024 + lane * 32);
#pragma unroll
    for (int n = 0; n < 16; n += 4) {
      const char* bp = smem + LDS_B1 + n * 1024 + lane * 32;
      v16bf b0 = *(const v16bf*)(bp);
      v16bf b1v = *(const v16bf*)(bp + 1024);
      v16bf b2v = *(const v16bf*)(bp + 2048);
      v16bf b3 = *(const v16bf*)(bp + 3072);
      acc[n + 0] = WMMA_BF16(a, b0,  acc[n + 0]);
      acc[n + 1] = WMMA_BF16(a, b1v, acc[n + 1]);
      acc[n + 2] = WMMA_BF16(a, b2v, acc[n + 2]);
      acc[n + 3] = WMMA_BF16(a, b3,  acc[n + 3]);
    }
  }

  // ---- Epilogue 1: bias+ReLU; restage 16x256 h1 (bf16) as A-fragments ----
  __syncthreads();   // all waves done reading stage-1 LDS
#pragma unroll
  for (int n = 0; n < 16; ++n) {
    int kcol = n * 16 + (lane & 15);      // h1 column == stage-2 K index
    float bias = b1[kcol];
#pragma unroll
    for (int r = 0; r < 8; ++r) {
      float v = acc[n][r] + bias;
      v = v > 0.f ? v : 0.f;
      *(unsigned short*)(smem + wave * 8192 + (kcol >> 5) * 1024 +
                         afrag_off(r + hi8, kcol & 31)) = f2bf(v);
    }
  }
  __syncthreads();

  // ---- Stage 2: out = h1 @ W2 + b2 (K=256, 4 output col-tiles) ----
  v8f acc2[4];
#pragma unroll
  for (int n = 0; n < 4; ++n) acc2[n] = vzero;
  for (int kk = 0; kk < HID / 32; ++kk) {
    const v16bf a = *(const v16bf*)(smem + wave * 8192 + kk * 1024 + lane * 32);
    const char* bp = smem + LDS_W2 + kk * 4096 + lane * 32;
    v16bf b0 = *(const v16bf*)(bp);
    v16bf b1v = *(const v16bf*)(bp + 1024);
    v16bf b2v = *(const v16bf*)(bp + 2048);
    v16bf b3 = *(const v16bf*)(bp + 3072);
    acc2[0] = WMMA_BF16(a, b0,  acc2[0]);
    acc2[1] = WMMA_BF16(a, b1v, acc2[1]);
    acc2[2] = WMMA_BF16(a, b2v, acc2[2]);
    acc2[3] = WMMA_BF16(a, b3,  acc2[3]);
  }
#pragma unroll
  for (int n = 0; n < 4; ++n) {
    int c = n * 16 + (lane & 15);
    float bias = b2[c];
#pragma unroll
    for (int r = 0; r < 8; ++r) {
      int gr = row0 + wave * 16 + r + hi8;
      if (gr < N) {
        float v = acc2[n][r] + bias;
        h[(size_t)gr * COUT + c] = v;
        z[(size_t)gr * COUT + c] = v;    // APPNP starts at z0 = h
      }
    }
  }
}

// ---------------- propagation / norm kernels ----------------
__global__ void fill_ones_kernel(float* __restrict__ deg, int N) {
  int i = blockIdx.x * blockDim.x + threadIdx.x;
  if (i < N) deg[i] = 1.0f;                       // self-loop contribution
}

__global__ void deg_accum_kernel(const int* __restrict__ dst, float* __restrict__ deg, int E) {
  int e = blockIdx.x * blockDim.x + threadIdx.x;
  if (e < E)
    __hip_atomic_fetch_add(&deg[dst[e]], 1.0f, __ATOMIC_RELAXED, __HIP_MEMORY_SCOPE_AGENT);
}

__global__ void deg_inv_kernel(const float* __restrict__ deg, float* __restrict__ dinv, int N) {
  int i = blockIdx.x * blockDim.x + threadIdx.x;
  if (i < N) dinv[i] = rsqrtf(deg[i]);            // deg >= 1 always
}

__global__ void self_init_kernel(const float* __restrict__ z, const float* __restrict__ dinv,
                                 float* __restrict__ agg, int total) {
  int t = blockIdx.x * blockDim.x + threadIdx.x;
  if (t < total) {
    float di = dinv[t >> 6];
    agg[t] = z[t] * di * di;                      // self-loop message, also inits agg
  }
}

__global__ void edge_scatter_kernel(const int* __restrict__ src, const int* __restrict__ dst,
                                    const float* __restrict__ dinv, const float* __restrict__ z,
                                    float* __restrict__ agg, int total) {
  int t = blockIdx.x * blockDim.x + threadIdx.x;  // E*64 = 204.8M < 2^31
  if (t >= total) return;
  int e = t >> 6, c = t & 63;
  int s = src[e], d = dst[e];
  float w = dinv[s] * dinv[d];
  __hip_atomic_fetch_add(&agg[(size_t)d * COUT + c], z[(size_t)s * COUT + c] * w,
                         __ATOMIC_RELAXED, __HIP_MEMORY_SCOPE_AGENT);
}

__global__ void combine_kernel(const float* __restrict__ agg, const float* __restrict__ h,
                               float* __restrict__ z, int total) {
  int t = blockIdx.x * blockDim.x + threadIdx.x;
  if (t < total) z[t] = 0.9f * agg[t] + 0.1f * h[t];
}

__global__ void __launch_bounds__(256)
logsoftmax_kernel(const float* __restrict__ z, float* __restrict__ out, int N) {
  int wave = threadIdx.x >> 5, lane = threadIdx.x & 31;
  int row = blockIdx.x * 8 + wave;
  if (row >= N) return;
  const float2 v = *(const float2*)(z + (size_t)row * COUT + lane * 2);
  float m = fmaxf(v.x, v.y);
  for (int off = 16; off; off >>= 1) m = fmaxf(m, __shfl_xor(m, off, 32));
  float s = __expf(v.x - m) + __expf(v.y - m);
  for (int off = 16; off; off >>= 1) s += __shfl_xor(s, off, 32);
  float ls = __logf(s);
  float2 o; o.x = v.x - m - ls; o.y = v.y - m - ls;
  *(float2*)(out + (size_t)row * COUT + lane * 2) = o;
}

extern "C" void kernel_launch(void* const* d_in, const int* in_sizes, int n_in,
                              void* d_out, int out_size, void* d_ws, size_t ws_size,
                              hipStream_t stream) {
  const float* x  = (const float*)d_in[0];
  const int*   ei = (const int*)d_in[1];
  const float* W1 = (const float*)d_in[2];
  const float* b1 = (const float*)d_in[3];
  const float* W2 = (const float*)d_in[4];
  const float* b2 = (const float*)d_in[5];
  float* out = (float*)d_out;

  const int N = in_sizes[0] / N_FEAT;   // 100000
  const int E = in_sizes[1] / 2;        // 3200000
  const int* src = ei;
  const int* dst = ei + E;

  float* h    = (float*)d_ws;
  float* z    = h    + (size_t)N * COUT;
  float* agg  = z    + (size_t)N * COUT;
  float* deg  = agg  + (size_t)N * COUT;
  float* dinv = deg  + N;

  // 1) Fused MLP encoder via bf16 WMMA; writes h and z0 = h.
  int gridEnc = (N + ROWS_PER_BLOCK - 1) / ROWS_PER_BLOCK;
  encoder_kernel<<<gridEnc, 128, LDS_SZ, stream>>>(x, W1, b1, W2, b2, h, z, N);

  // 2) gcn_norm: deg = 1 (self loop) + scatter-count(dst); dinv = rsqrt(deg).
  fill_ones_kernel<<<(N + 255) / 256, 256, 0, stream>>>(deg, N);
  deg_accum_kernel<<<(E + 255) / 256, 256, 0, stream>>>(dst, deg, E);
  deg_inv_kernel<<<(N + 255) / 256, 256, 0, stream>>>(deg, dinv, N);

  // 3) APPNP: z <- 0.9 * A_hat z + 0.1 * h, 10 hops. All buffers L2-resident.
  const int totN = N * COUT;
  const int totE = E * COUT;
  for (int hop = 0; hop < 10; ++hop) {
    self_init_kernel<<<(totN + 255) / 256, 256, 0, stream>>>(z, dinv, agg, totN);
    edge_scatter_kernel<<<(totE + 255) / 256, 256, 0, stream>>>(src, dst, dinv, z, agg, totE);
    combine_kernel<<<(totN + 255) / 256, 256, 0, stream>>>(agg, h, z, totN);
  }

  // 4) log_softmax over 64 channels, one wave32 per row.
  logsoftmax_kernel<<<(N + 7) / 8, 256, 0, stream>>>(z, out, N);
}
